// NonLocalBlock_45054206935466
// MI455X (gfx1250) — matrix operand order repaired and checked by reading, not compiled, not run on hardware
//
#include <hip/hip_runtime.h>

#define BN 4
#define CC 512
#define NN 4096
#define DD 256
#define BND ((size_t)BN * NN * DD)

typedef __attribute__((ext_vector_type(16))) _Float16 v16h;
typedef __attribute__((ext_vector_type(8)))  _Float16 v8h;
typedef __attribute__((ext_vector_type(8)))  float    v8f;
typedef __attribute__((ext_vector_type(4)))  unsigned int u32x4;
typedef __attribute__((ext_vector_type(8)))  int i32x8;
typedef __attribute__((ext_vector_type(4)))  int i32x4;

#if defined(__has_builtin)
#if __has_builtin(__builtin_amdgcn_tensor_load_to_lds)
#define HAVE_TDM 1
#endif
#endif
#if defined(__has_include)
#if __has_include(<hip/amd_detail/amd_gfx1250_TDM.h>)
#define TDM_6ARG 1
#endif
#endif

// ---------------------------------------------------------------------------
// TDM: one-shot DMA of a 64x256 f16 tile (rows contiguous in global) into
// LDS with a 264-half row stride via D# padding: every 128 DWORDs (one
// 256-half row) insert 4 DWORDs (8 halves).  D# per CDNA5 ISA §8.3/8.4:
// group0 = {count=1 | lds_addr | global_addr | type=2}, group1 holds
// data_size/pad/tensor dims/tile dims/stride.  Issued by one wave; tracked
// by TENSORcnt.
// ---------------------------------------------------------------------------
#ifdef HAVE_TDM
__device__ __forceinline__ void tdm_load_tile64x256(const void* gsrc,
                                                    void* ldsdst) {
  unsigned long long ga = (unsigned long long)(size_t)gsrc;
  unsigned lds_off = (unsigned)(size_t)ldsdst; // flat LDS addr low 32 = offset
  u32x4 g0;
  g0[0] = 1u;                                            // count=1 user D#
  g0[1] = lds_off;                                       // lds_addr
  g0[2] = (unsigned)ga;                                  // global_addr[31:0]
  g0[3] = (unsigned)((ga >> 32) & 0x01ffffffu) | (2u << 30); // [56:32], type=2
  i32x8 g1;
  g1[0] = (1 << 16)        // data_size = 1 -> 2 bytes
        | (1 << 20)        // pad_enable
        | (6 << 22)        // pad_interval: 2^(6+1)=128 DWORDs (one 256h row)
        | (3 << 25);       // pad_amount: 4 DWORDs (8 halves)
  g1[1] = 256 << 16;       // tensor_dim0[15:0] = 256
  g1[2] = (64 << 16);      // tensor_dim0[31:16]=0, tensor_dim1[15:0] = 64
  g1[3] = 256 << 16;       // tensor_dim1[31:16]=0, tile_dim0 = 256
  g1[4] = 64;              // tile_dim1 = 64, tile_dim2 = 0
  g1[5] = 256;             // tensor_dim0_stride[31:0] = 256 elements
  g1[6] = 0;
  g1[7] = 0;
  i32x4 z = (i32x4)0;
#ifdef TDM_6ARG
  __builtin_amdgcn_tensor_load_to_lds(g0, g1, z, z, (i32x8)0, 0);
#else
  __builtin_amdgcn_tensor_load_to_lds(g0, g1, z, z, 0);
#endif
}
#endif

// ---------------------------------------------------------------------------
// WMMA fragment helpers (wave32, 16x16x32 f16 -> f32)
//
// A-matrix 16x32 (MxK): lane l holds row M = l&15,
//   halves 0..7  -> K = kbase + 8*(l>>4) + i        (16B run)
//   halves 8..15 -> K = kbase + 16 + 8*(l>>4) + i   (16B run)
// B-matrix 32x16 (KxN): lane l holds col N = l&15,
//   same K mapping as A (halves 0..7 -> kbase+8*(l>>4)+i, 8..15 -> +16)
// C/D 16x16 f32: lane l holds col N = l&15, VGPR v -> row M = v + 8*(l>>4)
// ---------------------------------------------------------------------------
__device__ __forceinline__ v16h frag_a(const _Float16* base, int stride,
                                       int rbase, int kbase, int lane) {
  int r = rbase + (lane & 15);
  int g = lane >> 4;
  const _Float16* p = base + r * stride + kbase + 8 * g;
  v8h lo = *(const v8h*)p;
  v8h hi = *(const v8h*)(p + 16);
  v16h out;
#pragma unroll
  for (int i = 0; i < 8; ++i) { out[i] = lo[i]; out[i + 8] = hi[i]; }
  return out;
}

__device__ __forceinline__ v16h frag_b(const _Float16* base, int stride,
                                       int cbase, int kbase, int lane) {
  int c = cbase + (lane & 15);
  const _Float16* p = base + c * stride + kbase + 8 * (lane >> 4);
  v8h lo = *(const v8h*)p;
  v8h hi = *(const v8h*)(p + 16);
  v16h out;
#pragma unroll
  for (int i = 0; i < 8; ++i) { out[i] = lo[i]; out[i + 8] = hi[i]; }
  return out;
}

__device__ __forceinline__ v8f wmma_f16(v16h a, v16h b, v8f c) {
  return __builtin_amdgcn_wmma_f32_16x16x32_f16(false, a, false, b,
                                                (short)0, c, false, false);
}

__device__ __forceinline__ unsigned int h2u(_Float16 h) {
  union { _Float16 h; unsigned short u; } x;
  x.h = h;
  return (unsigned int)x.u;
}

// ---------------------------------------------------------------------------
// k0: convert the four f32 weight matrices to f16 in workspace.
// ---------------------------------------------------------------------------
__global__ void wconv_kernel(const float* __restrict__ tw,
                             const float* __restrict__ pw,
                             const float* __restrict__ gw,
                             const float* __restrict__ ow,
                             _Float16* __restrict__ dst) {
  int i = blockIdx.x * blockDim.x + threadIdx.x;
  if (i < CC * DD) {
    dst[i]               = (_Float16)tw[i];
    dst[CC * DD + i]     = (_Float16)pw[i];
    dst[2 * CC * DD + i] = (_Float16)gw[i];
    dst[3 * CC * DD + i] = (_Float16)ow[i];
  }
}

// ---------------------------------------------------------------------------
// k1: QKV projection.  out[o][b][n][d] = sum_c x[b][c][n] * w_o[d][c] + bias.
// Explicit ping-pong double buffering of {A, B0..B3}.
// ---------------------------------------------------------------------------
__global__ void proj_kernel(const float* __restrict__ x,
                            const _Float16* __restrict__ wts, // theta,phi,g
                            const float* __restrict__ tb,
                            const float* __restrict__ pb,
                            const float* __restrict__ gb,
                            _Float16* __restrict__ qkv) {
  extern __shared__ _Float16 sX[]; // [64][520]  x tile transposed: [n][c]
  const int nt = blockIdx.x, b = blockIdx.y;
  const int n0 = nt * 64;
  const int tid = threadIdx.x;
  const float* xb = x + (size_t)b * CC * NN;
  {
    int n = tid & 63, cstart = tid >> 6;
    for (int c = cstart; c < CC; c += 2)
      sX[n * 520 + c] = (_Float16)xb[(size_t)c * NN + n0 + n];
  }

  const int w = tid >> 5, lane = tid & 31, ln = lane & 15, g = lane >> 4;
  const float* biases[3] = {tb, pb, gb};
  for (int o = 0; o < 3; ++o) {
    const _Float16* wp = wts + (size_t)o * CC * DD; // [D][C]
    const float* bp = biases[o];
    _Float16* dst = qkv + (size_t)o * BND + ((size_t)b * NN + n0) * DD;
    for (int dtb = 0; dtb < 4; ++dtb) {      // 4 output d-tiles at a time
      __syncthreads();                        // fence: bound fragment liveness
      const int db = dtb * 64;
      v8f acc[4] = {};
      v16h a0  = frag_a(sX, 520, 16 * w, 0, lane);
      v16h p00 = frag_b(wp, CC, db +  0, 0, lane);
      v16h p01 = frag_b(wp, CC, db + 16, 0, lane);
      v16h p02 = frag_b(wp, CC, db + 32, 0, lane);
      v16h p03 = frag_b(wp, CC, db + 48, 0, lane);
#pragma unroll 1
      for (int ks2 = 0; ks2 < 7; ++ks2) {    // k-steps 0..13 (pairs)
        int k1 = (2 * ks2 + 1) * 32, k2 = (2 * ks2 + 2) * 32;
        v16h a1  = frag_a(sX, 520, 16 * w, k1, lane);
        v16h p10 = frag_b(wp, CC, db +  0, k1, lane);
        v16h p11 = frag_b(wp, CC, db + 16, k1, lane);
        v16h p12 = frag_b(wp, CC, db + 32, k1, lane);
        v16h p13 = frag_b(wp, CC, db + 48, k1, lane);
        acc[0] = wmma_f16(a0, p00, acc[0]);
        acc[1] = wmma_f16(a0, p01, acc[1]);
        acc[2] = wmma_f16(a0, p02, acc[2]);
        acc[3] = wmma_f16(a0, p03, acc[3]);
        a0  = frag_a(sX, 520, 16 * w, k2, lane);
        p00 = frag_b(wp, CC, db +  0, k2, lane);
        p01 = frag_b(wp, CC, db + 16, k2, lane);
        p02 = frag_b(wp, CC, db + 32, k2, lane);
        p03 = frag_b(wp, CC, db + 48, k2, lane);
        acc[0] = wmma_f16(a1, p10, acc[0]);
        acc[1] = wmma_f16(a1, p11, acc[1]);
        acc[2] = wmma_f16(a1, p12, acc[2]);
        acc[3] = wmma_f16(a1, p13, acc[3]);
      }
      { // tail: k-steps 14 (stage 0 regs) and 15
        int k1 = 15 * 32;
        v16h a1  = frag_a(sX, 520, 16 * w, k1, lane);
        v16h p10 = frag_b(wp, CC, db +  0, k1, lane);
        v16h p11 = frag_b(wp, CC, db + 16, k1, lane);
        v16h p12 = frag_b(wp, CC, db + 32, k1, lane);
        v16h p13 = frag_b(wp, CC, db + 48, k1, lane);
        acc[0] = wmma_f16(a0, p00, acc[0]);
        acc[1] = wmma_f16(a0, p01, acc[1]);
        acc[2] = wmma_f16(a0, p02, acc[2]);
        acc[3] = wmma_f16(a0, p03, acc[3]);
        acc[0] = wmma_f16(a1, p10, acc[0]);
        acc[1] = wmma_f16(a1, p11, acc[1]);
        acc[2] = wmma_f16(a1, p12, acc[2]);
        acc[3] = wmma_f16(a1, p13, acc[3]);
      }
#pragma unroll
      for (int j = 0; j < 4; ++j) {
        int col = (dtb * 4 + j) * 16 + ln;
        float bias = bp[col];
#pragma unroll
        for (int v = 0; v < 8; ++v) {
          int row = 16 * w + v + 8 * g;
          dst[(size_t)row * DD + col] = (_Float16)(acc[j][v] + bias);
        }
      }
    }
  }
}

// ---------------------------------------------------------------------------
// k2: flash attention.  Q and K tiles staged by the Tensor Data Mover when
// available (one DMA per tile, padded LDS rows, TENSORcnt-tracked); V staged
// transposed manually (TDM cannot transpose 16-bit).  S = Q K^T, online
// softmax (shfl_xor + v_exp_f32), P -> LDS f16, O += P V (ping-pong).
// ---------------------------------------------------------------------------
__global__ void attn_kernel(const _Float16* __restrict__ qkv,
                            _Float16* __restrict__ Y) {
  extern __shared__ _Float16 smem[];
  _Float16* sQ  = smem;              // [64][264]
  _Float16* sK  = sQ + 64 * 264;     // [64][264]
  _Float16* sVT = sK + 64 * 264;     // [256][72]  (V transposed: [d][m])
  _Float16* sP  = sVT + 256 * 72;    // [64][72]

  const int nt = blockIdx.x, b = blockIdx.y;
  const int n0 = nt * 64;
  const int tid = threadIdx.x;
  const int w = tid >> 5, lane = tid & 31, ln = lane & 15, g = lane >> 4;
  const _Float16* Qg = qkv + ((size_t)b * NN + n0) * DD;
  const _Float16* Kg = qkv + BND + (size_t)b * NN * DD;
  const _Float16* Vg = qkv + 2 * BND + (size_t)b * NN * DD;

#ifdef HAVE_TDM
  if (w == 0) tdm_load_tile64x256(Qg, sQ);
#else
  for (int p = tid * 8; p < 64 * DD; p += 1024) {
    int r = p >> 8, c = p & 255;
    *(v8h*)(sQ + r * 264 + c) = *(const v8h*)(Qg + p);
  }
#endif

  v8f o[16] = {};
  float rmax[8], rsum[8];
#pragma unroll
  for (int v = 0; v < 8; ++v) { rmax[v] = -1e30f; rsum[v] = 0.f; }

  for (int mt = 0; mt < NN / 64; ++mt) {
    const size_t m0 = (size_t)mt * 64 * DD;
    // stage K tile [m][d]
#ifdef HAVE_TDM
    if (w == 0) tdm_load_tile64x256(Kg + m0, sK);
#else
    for (int p = tid * 8; p < 64 * DD; p += 1024) {
      int r = p >> 8, c = p & 255;
      *(v8h*)(sK + r * 264 + c) = *(const v8h*)(Kg + m0 + p);
    }
#endif
    // stage V tile transposed [d][m]: pack 4 consecutive rows (m) into one
    // ds_store_b64 (m is the contiguous axis of sVT)
    for (int ch = tid; ch < 16 * 32; ch += 128) {
      int mq = ch & 15;        // m-quad: rows 4*mq .. 4*mq+3
      int dq = ch >> 4;        // d-octet: cols 8*dq .. 8*dq+7
      int dbase = dq * 8;
      const _Float16* src = Vg + m0 + (size_t)(4 * mq) * DD + dbase;
      v8h r0 = *(const v8h*)src;
      v8h r1 = *(const v8h*)(src + DD);
      v8h r2 = *(const v8h*)(src + 2 * DD);
      v8h r3 = *(const v8h*)(src + 3 * DD);
#pragma unroll
      for (int j = 0; j < 8; ++j) {
        uint2 pk;
        pk.x = h2u(r0[j]) | (h2u(r1[j]) << 16);
        pk.y = h2u(r2[j]) | (h2u(r3[j]) << 16);
        *(uint2*)(sVT + (dbase + j) * 72 + 4 * mq) = pk;
      }
    }
#ifdef HAVE_TDM
    if (w == 0) __builtin_amdgcn_s_wait_tensorcnt(0);  // drain Q/K DMAs
#endif
    __syncthreads();

    // prefetch next K/V tile while we compute on this one
    if (mt + 1 < NN / 64) {
      const size_t m1 = m0 + (size_t)64 * DD;
      __builtin_prefetch(Kg + m1 + (size_t)tid * 128, 0, 1);
      __builtin_prefetch(Vg + m1 + (size_t)tid * 128, 0, 1);
    }

    // S = Q * K^T  (16 rows x 64 keys); one A feeds 4 named B-fragments
    v8f s[4] = {};
#pragma unroll 1
    for (int ks = 0; ks < 8; ++ks) {
      v16h a  = frag_a(sQ, 264, 16 * w, ks * 32, lane);
      v16h b0 = frag_b(sK, 264,  0, ks * 32, lane);
      v16h b1 = frag_b(sK, 264, 16, ks * 32, lane);
      v16h b2 = frag_b(sK, 264, 32, ks * 32, lane);
      v16h b3 = frag_b(sK, 264, 48, ks * 32, lane);
      s[0] = wmma_f16(a, b0, s[0]);
      s[1] = wmma_f16(a, b1, s[1]);
      s[2] = wmma_f16(a, b2, s[2]);
      s[3] = wmma_f16(a, b3, s[3]);
    }

    // online softmax (row = v + 8g, spread over the 16 lanes of a half-wave)
    float corr[8];
#pragma unroll
    for (int v = 0; v < 8; ++v) {
      float mx = fmaxf(fmaxf(s[0][v], s[1][v]), fmaxf(s[2][v], s[3][v]));
      mx = fmaxf(mx, __shfl_xor(mx, 1, 32));
      mx = fmaxf(mx, __shfl_xor(mx, 2, 32));
      mx = fmaxf(mx, __shfl_xor(mx, 4, 32));
      mx = fmaxf(mx, __shfl_xor(mx, 8, 32));
      float mnew = fmaxf(rmax[v], mx);
      float c = __expf(rmax[v] - mnew);
      rmax[v] = mnew;
      float rs = 0.f;
#pragma unroll
      for (int ct = 0; ct < 4; ++ct) {
        float e = __expf(s[ct][v] - mnew);
        s[ct][v] = e;
        rs += e;
      }
      rs += __shfl_xor(rs, 1, 32);
      rs += __shfl_xor(rs, 2, 32);
      rs += __shfl_xor(rs, 4, 32);
      rs += __shfl_xor(rs, 8, 32);
      rsum[v] = rsum[v] * c + rs;
      corr[v] = c;
    }
#pragma unroll
    for (int dt = 0; dt < 16; ++dt)
#pragma unroll
      for (int v = 0; v < 8; ++v) o[dt][v] *= corr[v];

    // P -> LDS (f16, A-layout rows); each wave reads only rows it wrote
#pragma unroll
    for (int ct = 0; ct < 4; ++ct)
#pragma unroll
      for (int v = 0; v < 8; ++v)
        sP[(16 * w + v + 8 * g) * 72 + ct * 16 + ln] = (_Float16)s[ct][v];

    // O += P * V  (ping-pong pipelined B-fragments, fully unrolled)
    {
      v16h pa0 = frag_a(sP, 72, 16 * w, 0, lane);
      v16h pa1 = frag_a(sP, 72, 16 * w, 32, lane);
      v16h q00 = frag_b(sVT, 72, 0, 0, lane);
      v16h q01 = frag_b(sVT, 72, 0, 32, lane);
#pragma unroll
      for (int dt2 = 0; dt2 < 7; ++dt2) {    // d-tiles 0..13 (pairs)
        int d1 = (2 * dt2 + 1) * 16, d2 = (2 * dt2 + 2) * 16;
        v16h q10 = frag_b(sVT, 72, d1, 0, lane);
        v16h q11 = frag_b(sVT, 72, d1, 32, lane);
        o[2 * dt2]     = wmma_f16(pa0, q00, o[2 * dt2]);
        o[2 * dt2]     = wmma_f16(pa1, q01, o[2 * dt2]);
        q00 = frag_b(sVT, 72, d2, 0, lane);
        q01 = frag_b(sVT, 72, d2, 32, lane);
        o[2 * dt2 + 1] = wmma_f16(pa0, q10, o[2 * dt2 + 1]);
        o[2 * dt2 + 1] = wmma_f16(pa1, q11, o[2 * dt2 + 1]);
      }
      { // tail: d-tiles 14 (stage 0 regs) and 15
        v16h q10 = frag_b(sVT, 72, 15 * 16, 0, lane);
        v16h q11 = frag_b(sVT, 72, 15 * 16, 32, lane);
        o[14] = wmma_f16(pa0, q00, o[14]);
        o[14] = wmma_f16(pa1, q01, o[14]);
        o[15] = wmma_f16(pa0, q10, o[15]);
        o[15] = wmma_f16(pa1, q11, o[15]);
      }
    }
    __syncthreads();
  }

  _Float16* Yb = Y + ((size_t)b * NN + n0) * DD;
#pragma unroll
  for (int v = 0; v < 8; ++v) rsum[v] = 1.0f / rsum[v];
#pragma unroll
  for (int dt = 0; dt < 16; ++dt) {
    int col = dt * 16 + ln;
#pragma unroll
    for (int v = 0; v < 8; ++v) {
      int row = 16 * w + v + 8 * g;
      Yb[(size_t)row * DD + col] = (_Float16)(o[dt][v] * rsum[v]);
    }
  }
}

// ---------------------------------------------------------------------------
// k3: out[b][c][n] = sum_d out_w[c][d] * Y[b][n][d] + out_b[c] + x[b][c][n].
// Y tile staged by TDM when available; explicit ping-pong double buffering
// (global A + 4 LDS B fragments).
// ---------------------------------------------------------------------------
__global__ void outproj_kernel(const float* __restrict__ x,
                               const _Float16* __restrict__ wO, // [C][D] f16
                               const float* __restrict__ ob,
                               const _Float16* __restrict__ Y,
                               float* __restrict__ out) {
  extern __shared__ _Float16 sY[]; // [64][264]
  const int nt = blockIdx.x, b = blockIdx.y;
  const int n0 = nt * 64;
  const int tid = threadIdx.x;
  const int w = tid >> 5, lane = tid & 31, ln = lane & 15, g = lane >> 4;
  const _Float16* Yg = Y + ((size_t)b * NN + n0) * DD;
#ifdef HAVE_TDM
  if (w == 0) {
    tdm_load_tile64x256(Yg, sY);
    __builtin_amdgcn_s_wait_tensorcnt(0);
  }
#else
  for (int p = tid * 8; p < 64 * DD; p += 1024) {
    int r = p >> 8, c = p & 255;
    *(v8h*)(sY + r * 264 + c) = *(const v8h*)(Yg + p);
  }
#endif

  for (int ct = 0; ct < 8; ++ct) {
    int c0 = 128 * w + 16 * ct;
    __syncthreads();                          // fence: bound fragment liveness
    v8f acc[4] = {};
    v16h a0  = frag_a(wO, DD, c0, 0, lane);
    v16h p00 = frag_b(sY, 264,  0, 0, lane);
    v16h p01 = frag_b(sY, 264, 16, 0, lane);
    v16h p02 = frag_b(sY, 264, 32, 0, lane);
    v16h p03 = frag_b(sY, 264, 48, 0, lane);
#pragma unroll 1
    for (int ks2 = 0; ks2 < 3; ++ks2) {      // k-steps 0..5 (pairs)
      int k1 = (2 * ks2 + 1) * 32, k2 = (2 * ks2 + 2) * 32;
      v16h a1  = frag_a(wO, DD, c0, k1, lane);
      v16h p10 = frag_b(sY, 264,  0, k1, lane);
      v16h p11 = frag_b(sY, 264, 16, k1, lane);
      v16h p12 = frag_b(sY, 264, 32, k1, lane);
      v16h p13 = frag_b(sY, 264, 48, k1, lane);
      acc[0] = wmma_f16(a0, p00, acc[0]);
      acc[1] = wmma_f16(a0, p01, acc[1]);
      acc[2] = wmma_f16(a0, p02, acc[2]);
      acc[3] = wmma_f16(a0, p03, acc[3]);
      a0  = frag_a(wO, DD, c0, k2, lane);
      p00 = frag_b(sY, 264,  0, k2, lane);
      p01 = frag_b(sY, 264, 16, k2, lane);
      p02 = frag_b(sY, 264, 32, k2, lane);
      p03 = frag_b(sY, 264, 48, k2, lane);
      acc[0] = wmma_f16(a1, p10, acc[0]);
      acc[1] = wmma_f16(a1, p11, acc[1]);
      acc[2] = wmma_f16(a1, p12, acc[2]);
      acc[3] = wmma_f16(a1, p13, acc[3]);
    }
    { // tail: k-steps 6 (stage 0 regs) and 7
      int k1 = 7 * 32;
      v16h a1  = frag_a(wO, DD, c0, k1, lane);
      v16h p10 = frag_b(sY, 264,  0, k1, lane);
      v16h p11 = frag_b(sY, 264, 16, k1, lane);
      v16h p12 = frag_b(sY, 264, 32, k1, lane);
      v16h p13 = frag_b(sY, 264, 48, k1, lane);
      acc[0] = wmma_f16(a0, p00, acc[0]);
      acc[1] = wmma_f16(a0, p01, acc[1]);
      acc[2] = wmma_f16(a0, p02, acc[2]);
      acc[3] = wmma_f16(a0, p03, acc[3]);
      acc[0] = wmma_f16(a1, p10, acc[0]);
      acc[1] = wmma_f16(a1, p11, acc[1]);
      acc[2] = wmma_f16(a1, p12, acc[2]);
      acc[3] = wmma_f16(a1, p13, acc[3]);
    }
#pragma unroll
    for (int ntile = 0; ntile < 4; ++ntile) {
      int ncol = n0 + ntile * 16 + ln;
#pragma unroll
      for (int v = 0; v < 8; ++v) {
        int crow = c0 + v + 8 * g;
        size_t idx = ((size_t)b * CC + crow) * NN + ncol;
        out[idx] = acc[ntile][v] + ob[crow] + x[idx];
      }
    }
  }
}

// ---------------------------------------------------------------------------
extern "C" void kernel_launch(void* const* d_in, const int* in_sizes, int n_in,
                              void* d_out, int out_size, void* d_ws, size_t ws_size,
                              hipStream_t stream) {
  const float* x  = (const float*)d_in[0];
  const float* tw = (const float*)d_in[1];
  const float* tb = (const float*)d_in[2];
  const float* pw = (const float*)d_in[3];
  const float* pb = (const float*)d_in[4];
  const float* gw = (const float*)d_in[5];
  const float* gb = (const float*)d_in[6];
  const float* ow = (const float*)d_in[7];
  const float* ob = (const float*)d_in[8];
  float* out = (float*)d_out;

  _Float16* ws = (_Float16*)d_ws;
  _Float16* W  = ws;                    // 4 * 131072 halves (theta,phi,g,out_w)
  _Float16* Q  = ws + 4 * CC * DD;      // Q, K, V each BND halves
  _Float16* Y  = Q + 3 * BND;           // BND halves

  wconv_kernel<<<dim3((CC * DD + 255) / 256), 256, 0, stream>>>(tw, pw, gw, ow, W);

  size_t sh_proj = (size_t)64 * 520 * sizeof(_Float16);                       // 66560 B
  proj_kernel<<<dim3(NN / 64, BN), 128, sh_proj, stream>>>(x, W, tb, pb, gb, Q);

  size_t sh_attn = (size_t)(64 * 264 + 64 * 264 + 256 * 72 + 64 * 72) * sizeof(_Float16); // 113664 B
  attn_kernel<<<dim3(NN / 64, BN), 128, sh_attn, stream>>>(Q, Y);

  size_t sh_out = (size_t)64 * 264 * sizeof(_Float16);                        // 33792 B
  outproj_kernel<<<dim3(NN / 64, BN), 128, sh_out, stream>>>(x, W + 3 * CC * DD, ob, Y, out);
}
